// CGCN_71038759076270
// MI455X (gfx1250) — matrix operand order repaired and checked by reading, not compiled, not run on hardware
//
#include <hip/hip_runtime.h>
#include <hip/hip_bf16.h>
#include <float.h>

typedef __attribute__((ext_vector_type(16))) _Float16 v16h;
typedef __attribute__((ext_vector_type(8)))  float    v8f;
typedef __attribute__((ext_vector_type(4)))  int      v4i;

#define DCOLS 64   // reference DC

#if defined(__HIP_DEVICE_COMPILE__) && \
    __has_builtin(__builtin_amdgcn_global_load_async_to_lds_b128) && \
    __has_builtin(__builtin_amdgcn_s_wait_asynccnt)
#define USE_ASYNC_LDS 1
#else
#define USE_ASYNC_LDS 0
#endif

// ---------- helpers ----------
__device__ __forceinline__ float warpSum(float v) {
#pragma unroll
  for (int off = 16; off > 0; off >>= 1) v += __shfl_xor(v, off, 32);
  return v;
}

// order-preserving float<->uint map so atomicMax(u32) implements float max
__device__ __forceinline__ unsigned fOrd(float f) {
  unsigned u = __float_as_uint(f);
  return (u >> 31) ? ~u : (u | 0x80000000u);
}
__device__ __forceinline__ float fUnord(unsigned e) {
  return (e >> 31) ? __uint_as_float(e & 0x7fffffffu) : __uint_as_float(~e);
}

#if USE_ASYNC_LDS
__device__ __forceinline__ void asyncLoadB128(const float* g, float* l) {
  // toolchain signature (from diagnostic): (v4i*, v4i*, imm offset, imm cpol)
  __builtin_amdgcn_global_load_async_to_lds_b128((v4i*)g, (v4i*)l, 0, 0);
}
#endif

// ---------- kernels ----------
// Repack w [DF,64] f32 -> f16 in WMMA-B fragment order:
// dst = ((tile*64 + col)*2 + half)*16 + j  where k = 32*tile + 8*half + j (j<8), 32*tile + 16 + 8*half + (j-8) (j>=8)
__global__ void cvt_w_kernel(const float* __restrict__ w, _Float16* __restrict__ wh, int n) {
  int i = blockIdx.x * blockDim.x + threadIdx.x;
  if (i >= n) return;
  int k = i >> 6, c = i & 63;
  int tile = k >> 5, kk = k & 31;
  int half = (kk >> 3) & 1, j = (kk & 7) | ((kk >> 4) << 3);
  size_t dst = (((size_t)tile * 64 + c) * 2 + half) * 16 + j;
  wh[dst] = (_Float16)w[i];
}

__global__ void zero_kernel(unsigned* __restrict__ p, long long n) {
  long long i = (long long)blockIdx.x * blockDim.x + threadIdx.x;
  if (i < n) p[i] = 0u;
}

// feat = leaky_relu(A @ W + b); A [NI,DF] f32, W packed f16 fragments.
// 4 waves/block; block = 16-row strip; wave w -> cols [16w,16w+16).
// A tile double-buffered in LDS via GLOBAL_LOAD_ASYNC_TO_LDS_B128 (ASYNCcnt).
__global__ __launch_bounds__(128)
void gemm_feat_kernel(const float* __restrict__ A, const _Float16* __restrict__ Bp,
                      const float* __restrict__ bias, float* __restrict__ out,
                      int NI, int DF) {
  __shared__ __align__(16) float As[2][16 * 36];   // 16 rows x 32 K, stride 36 (144B, 16B-aligned)
  const int tid  = threadIdx.x;
  const int wave = tid >> 5;
  const int lane = tid & 31;
  const int row0 = blockIdx.x * 16;
  const int col0 = wave * 16;
  const int m    = lane & 15;
  const int half = lane >> 4;

  // staging slot: thread tid -> row sr, 4 consecutive K at sk (128 thr * float4 = 512 floats)
  const int sr = tid >> 3;
  const int sk = (tid & 7) * 4;
  const int srow = min(row0 + sr, NI - 1);           // clamp: branch-free OOB handling
  const float* gsrc = A + (size_t)srow * DF + sk;
  float* ldst[2] = { &As[0][sr * 36 + sk], &As[1][sr * 36 + sk] };

  const int T = DF >> 5;                              // K tiles of 32
  v8f c = {};

#if USE_ASYNC_LDS
  asyncLoadB128(gsrc, ldst[0]);                       // prefetch tile 0
#endif
  for (int t = 0; t < T; ++t) {
#if USE_ASYNC_LDS
    if (t + 1 < T) {
      asyncLoadB128(gsrc + (size_t)(t + 1) * 32, ldst[(t + 1) & 1]);
      __builtin_amdgcn_s_wait_asynccnt(1);            // tile t landed; t+1 in flight
    } else {
      __builtin_amdgcn_s_wait_asynccnt(0);
    }
    __syncthreads();
    const float* buf = As[t & 1];
#else
    __syncthreads();
    *(float4*)ldst[0] = *(const float4*)(gsrc + (size_t)t * 32);
    __syncthreads();
    const float* buf = As[0];
#endif
    // A fragment (16-bit A 16x32 layout): runs [8h,8h+8) and [16+8h,16+8h+8)
    const float* ap = buf + m * 36 + half * 8;
    float4 a0 = *(const float4*)(ap);
    float4 a1 = *(const float4*)(ap + 4);
    float4 a2 = *(const float4*)(ap + 16);
    float4 a3 = *(const float4*)(ap + 20);
    v16h a;
    a[0]  = (_Float16)a0.x; a[1]  = (_Float16)a0.y; a[2]  = (_Float16)a0.z; a[3]  = (_Float16)a0.w;
    a[4]  = (_Float16)a1.x; a[5]  = (_Float16)a1.y; a[6]  = (_Float16)a1.z; a[7]  = (_Float16)a1.w;
    a[8]  = (_Float16)a2.x; a[9]  = (_Float16)a2.y; a[10] = (_Float16)a2.z; a[11] = (_Float16)a2.w;
    a[12] = (_Float16)a3.x; a[13] = (_Float16)a3.y; a[14] = (_Float16)a3.z; a[15] = (_Float16)a3.w;
    // B fragment: one contiguous 32B chunk per lane (packed by cvt_w_kernel)
    v16h b = *((const v16h*)Bp + (((size_t)t * 64 + col0 + m) * 2 + half));

    c = __builtin_amdgcn_wmma_f32_16x16x32_f16(false, a, false, b, (short)0, c, false, false);
#if USE_ASYNC_LDS
    __syncthreads();   // everyone done reading buf[t&1] before it is re-issued at t+2
#endif
  }
  // C layout: VGPR v, lanes0-15 -> M=v, lanes16-31 -> M=8+v; N = lane%16
  const float bb = bias[col0 + m];
#pragma unroll
  for (int v = 0; v < 8; ++v) {
    int r = row0 + half * 8 + v;
    if (r < NI) {
      float val = c[v] + bb;
      out[(size_t)r * DCOLS + col0 + m] = (val >= 0.0f) ? val : 0.01f * val;
    }
  }
}

// dst[row] = l2norm(src[row] (+ add[row])); one wave per 64-float row
__global__ void l2norm_kernel(const float* __restrict__ src, const float* __restrict__ add,
                              float* __restrict__ dst, int nrows) {
  int wave = threadIdx.x >> 5;
  int lane = threadIdx.x & 31;
  int row  = blockIdx.x * 8 + wave;
  if (row >= nrows) return;
  const float2* sp = (const float2*)(src + (size_t)row * DCOLS);
  float2 v = sp[lane];
  if (add) {
    const float2* ap = (const float2*)(add + (size_t)row * DCOLS);
    float2 w = ap[lane];
    v.x += w.x; v.y += w.y;
  }
  float ss = warpSum(v.x * v.x + v.y * v.y);
  float scale = 1.0f / fmaxf(sqrtf(ss), 1e-12f);
  float2* dp = (float2*)(dst + (size_t)row * DCOLS);
  dp[lane] = make_float2(v.x * scale, v.y * scale);
}

// per-edge cosine logit + segment max (wave per edge)
__global__ void edge_dot_kernel(const float* __restrict__ xn, const int* __restrict__ rows,
                                const int* __restrict__ cols, int E,
                                unsigned* __restrict__ mbuf, float* __restrict__ alpha) {
  int wave = threadIdx.x >> 5;
  int lane = threadIdx.x & 31;
  int e = blockIdx.x * 8 + wave;
  if (e >= E) return;
  int r = rows[e], s = cols[e];
  const float2* rp = (const float2*)(xn + (size_t)r * DCOLS);
  const float2* cp = (const float2*)(xn + (size_t)s * DCOLS);
  float2 a = rp[lane], b = cp[lane];
  float d = warpSum(a.x * b.x + a.y * b.y);
  if (lane == 0) {
    alpha[e] = d;
    atomicMax(&mbuf[r], fOrd(d));
  }
}

// exp(logit - seg_max) + segment sum (thread per edge)
__global__ void edge_exp_kernel(float* __restrict__ alpha, const int* __restrict__ rows,
                                int E, const unsigned* __restrict__ mbuf,
                                float* __restrict__ sbuf) {
  int e = blockIdx.x * blockDim.x + threadIdx.x;
  if (e >= E) return;
  int r = rows[e];
  float ex = expf(alpha[e] - fUnord(mbuf[r]));
  alpha[e] = ex;
  atomicAdd(&sbuf[r], ex);
}

// normalize alpha; xhat[dst] += alpha * x[src]  (wave per edge)
__global__ void edge_scatter_kernel(const float* __restrict__ x, const int* __restrict__ rows,
                                    const int* __restrict__ cols, int E,
                                    const float* __restrict__ sbuf,
                                    const float* __restrict__ alphaIn,
                                    float* __restrict__ xhat, float* __restrict__ alphaOut) {
  int wave = threadIdx.x >> 5;
  int lane = threadIdx.x & 31;
  int e = blockIdx.x * 8 + wave;
  if (e >= E) return;
  int r = rows[e], s = cols[e];
  float a = alphaIn[e] / (sbuf[r] + 1e-16f);
  if (lane == 0) alphaOut[e] = a;
  const float2* cp = (const float2*)(x + (size_t)s * DCOLS);
  float2 v = cp[lane];
  float* op = xhat + (size_t)r * DCOLS + lane * 2;
  atomicAdd(op,     a * v.x);
  atomicAdd(op + 1, a * v.y);
}

// y = x + leaky_relu(xhat)
__global__ void final_out_kernel(const float* __restrict__ x, const float* __restrict__ xhat,
                                 float* __restrict__ y, long long n) {
  long long i = (long long)blockIdx.x * blockDim.x + threadIdx.x;
  if (i < n) {
    float h = xhat[i];
    h = (h >= 0.0f) ? h : 0.01f * h;
    y[i] = x[i] + h;
  }
}

// ---------- launcher ----------
extern "C" void kernel_launch(void* const* d_in, const int* in_sizes, int n_in,
                              void* d_out, int out_size, void* d_ws, size_t ws_size,
                              hipStream_t stream) {
  const float* features   = (const float*)d_in[0];
  const float* mlp_w      = (const float*)d_in[1];
  const float* mlp_b      = (const float*)d_in[2];
  const float* preference = (const float*)d_in[3];
  const int*   adj        = (const int*)d_in[4];
  const int*   adj_user   = (const int*)d_in[5];

  const int DC = in_sizes[2];               // 64
  const int DF = in_sizes[1] / DC;          // 2048
  const int NI = in_sizes[0] / DF;          // 30000
  const int NU = in_sizes[3] / DC;          // 50000
  const int E2 = in_sizes[4] / 2;           // 2,000,000
  const int EU = in_sizes[5] / 2;           // 1,000,000
  const int N  = NU + NI;

  char* ws = (char*)d_ws;
  _Float16* Bh   = (_Float16*)ws;  ws += (size_t)DF * DC * sizeof(_Float16);
  float* x       = (float*)ws;     ws += (size_t)N * DC * sizeof(float);
  float* xn      = (float*)ws;     ws += (size_t)N * DC * sizeof(float);
  float* xhat    = (float*)ws;     ws += (size_t)N * DC * sizeof(float);   // zero-region start
  unsigned* mbuf = (unsigned*)ws;  ws += (size_t)N * sizeof(unsigned);     // contiguous after xhat
  float* sbuf    = (float*)ws;     ws += (size_t)N * sizeof(float);        // contiguous after mbuf
  float* alphaB  = (float*)ws;     ws += (size_t)E2 * sizeof(float);

  float* y = (float*)d_out;
  float* alphaFinal = y + (size_t)N * DC;

  const int* rows2 = adj;        const int* cols2 = adj + E2;
  const int* rowsU = adj_user;   const int* colsU = adj_user + EU;

  // feat = l2norm(leaky_relu(features @ W + b));  pref = l2norm(preference)
  cvt_w_kernel<<<(DF * DC + 255) / 256, 256, 0, stream>>>(mlp_w, Bh, DF * DC);
  gemm_feat_kernel<<<(NI + 15) / 16, 128, 0, stream>>>(features, Bh, mlp_b,
                                                       x + (size_t)NU * DC, NI, DF);
  l2norm_kernel<<<(NU + 7) / 8, 256, 0, stream>>>(preference, nullptr, x, NU);
  l2norm_kernel<<<(NI + 7) / 8, 256, 0, stream>>>(x + (size_t)NU * DC, nullptr,
                                                  x + (size_t)NU * DC, NI);

  const long long zn = (long long)N * DC + 2LL * N;   // xhat + mbuf + sbuf (all-zero bits)
  const unsigned zb  = (unsigned)((zn + 255) / 256);

  // 3 routing iterations on user<-item edges
  for (int it = 0; it < 3; ++it) {
    l2norm_kernel<<<(N + 7) / 8, 256, 0, stream>>>(x, nullptr, xn, N);
    zero_kernel<<<zb, 256, 0, stream>>>((unsigned*)xhat, zn);
    edge_dot_kernel<<<(EU + 7) / 8, 256, 0, stream>>>(xn, rowsU, colsU, EU, mbuf, alphaB);
    edge_exp_kernel<<<(EU + 255) / 256, 256, 0, stream>>>(alphaB, rowsU, EU, mbuf, sbuf);
    edge_scatter_kernel<<<(EU + 7) / 8, 256, 0, stream>>>(x, rowsU, colsU, EU, sbuf,
                                                          alphaB, xhat, alphaB);
    // pref = l2norm(pref + xhat[:NU])
    l2norm_kernel<<<(NU + 7) / 8, 256, 0, stream>>>(x, xhat, x, NU);
  }

  // final refine on the full symmetric edge list
  l2norm_kernel<<<(N + 7) / 8, 256, 0, stream>>>(x, nullptr, xn, N);
  zero_kernel<<<zb, 256, 0, stream>>>((unsigned*)xhat, zn);
  edge_dot_kernel<<<(E2 + 7) / 8, 256, 0, stream>>>(xn, rows2, cols2, E2, mbuf, alphaB);
  edge_exp_kernel<<<(E2 + 255) / 256, 256, 0, stream>>>(alphaB, rows2, E2, mbuf, sbuf);
  edge_scatter_kernel<<<(E2 + 7) / 8, 256, 0, stream>>>(x, rows2, cols2, E2, sbuf,
                                                        alphaB, xhat, alphaFinal);
  final_out_kernel<<<(unsigned)(((long long)N * DC + 255) / 256), 256, 0, stream>>>(
      x, xhat, y, (long long)N * DC);
}
